// Net_3006477107432
// MI455X (gfx1250) — compile-verified, matrix operand order checked by reading, not run on hardware
//
#include <hip/hip_runtime.h>
#include <hip/hip_bf16.h>

typedef __attribute__((ext_vector_type(16))) __bf16 v16bf;
typedef __attribute__((ext_vector_type(2)))  __bf16 v2bf;
typedef __attribute__((ext_vector_type(8)))  float  v8f;

#define NATOMS 64
#define NBATCH 2048
#define DIN    384
#define H1D    128
#define H2D    128
#define H3D    64

#if defined(__has_builtin)
#if __has_builtin(__builtin_amdgcn_tensor_load_to_lds) && \
    __has_builtin(__builtin_amdgcn_s_wait_tensorcnt)
#define HAVE_TDM 1
#endif
#endif

#ifdef HAVE_TDM
typedef unsigned int v4u  __attribute__((ext_vector_type(4)));
typedef int          v8i_ __attribute__((ext_vector_type(8)));
typedef int          v4i_ __attribute__((ext_vector_type(4)));

// TDM: async-load a 16-row x 64-col f32 tile (row stride = NATOMS*DIN elements)
// from global into LDS. D# bit packing per CDNA5 ISA ch.8 (§8.3/§8.4):
//  group0: w0 = count=1 ; w1 = lds_addr ; w2/w3 = 57-bit global addr ; type=2
//  group1: data_size=2 (4B) ; tensor_dim0=64 ; tensor_dim1=16 ;
//          tile_dim0=64 ; tile_dim1=16 ; tensor_dim0_stride=24576
__device__ __forceinline__ void tdm_load_x_tile(const float* gsrc, float* ldsDst) {
  unsigned long long ga = (unsigned long long)(size_t)gsrc;
  unsigned int lds = (unsigned int)(size_t)ldsDst;   // low 32 bits = LDS offset
  v4u g0 = { 1u,                                      // count = 1 descriptor
             lds,
             (unsigned int)ga,
             (unsigned int)((ga >> 32) & 0x01FFFFFFu) | (2u << 30) };
  v8i_ g1 = { (int)(2u << 16),                        // data_size = 4 bytes
              (int)(64u << 16),                       // tensor_dim0[15:0]
              (int)(16u << 16),                       // tensor_dim0 hi | tensor_dim1 lo
              (int)(64u << 16),                       // tensor_dim1 hi | tile_dim0
              16,                                     // tile_dim1 | tile_dim2<<16
              (int)(NATOMS * DIN),                    // tensor_dim0_stride lo32
              0, 0 };
  v4i_ gz4 = { 0, 0, 0, 0 };
  v8i_ gz8 = { 0, 0, 0, 0, 0, 0, 0, 0 };
  __builtin_amdgcn_tensor_load_to_lds(g0, g1, gz4, gz4, gz8, 0);
}
#endif

// 16x32 bf16 WMMA fragment from row-major bf16 LDS tile (ISA 7.12.2 layout):
// lane half h = lane>>4 ; pair j holds k = kofs + 16*(j>>2) + 8*h + 2*(j&3), k+1
__device__ __forceinline__ v16bf load_frag_bf16(const __bf16* __restrict__ base,
                                                int row, int stride, int kofs, int h) {
  v16bf f;
#pragma unroll
  for (int j = 0; j < 8; ++j) {
    int k = kofs + ((j >> 2) << 4) + (h << 3) + ((j & 3) << 1);
    const __bf16* p = base + row * stride + k;
    f[2 * j]     = p[0];
    f[2 * j + 1] = p[1];
  }
  return f;
}

// Same fragment, sourced from f32 LDS (layer-0 x tile), converting to bf16.
__device__ __forceinline__ v16bf load_frag_f32(const float* __restrict__ base,
                                               int row, int stride, int kofs, int h) {
  v16bf f;
#pragma unroll
  for (int j = 0; j < 8; ++j) {
    int k = kofs + ((j >> 2) << 4) + (h << 3) + ((j & 3) << 1);
    const float* p = base + row * stride + k;
    f[2 * j]     = (__bf16)p[0];
    f[2 * j + 1] = (__bf16)p[1];
  }
  return f;
}

__global__ __launch_bounds__(256) void species_mlp_kernel(
    const float* __restrict__ x, const int* __restrict__ species,
    const float* __restrict__ W0, const float* __restrict__ b0,
    const float* __restrict__ W1, const float* __restrict__ b1,
    const float* __restrict__ W2, const float* __restrict__ b2,
    const float* __restrict__ W3, const float* __restrict__ b3,
    float* __restrict__ out)
{
  __shared__ __bf16 wbufT[H1D * 64];          // 16 KB transposed weight chunk [n][k]
  __shared__ __bf16 actS[8][16 * H1D];        // 8 waves x 4 KB act / x-stage tiles
  __shared__ float  b0s[H1D], b1s[H2D], b2s[H3D], w3s[H3D];
  __shared__ float  b3s;

  const int tid  = threadIdx.x;
  const int lane = tid & 31;
  const int wave = tid >> 5;
  const int h    = lane >> 4;      // lane half
  const int mr   = lane & 15;      // M row (A frag) == N col (B/C frag)

  const int atom  = blockIdx.x & (NATOMS - 1);
  const int btile = blockIdx.x >> 6;
  const int bRow0 = btile * 128 + wave * 16;

  const int s = species[atom];
  const float* W0g = W0 + (size_t)s * DIN * H1D;
  const float* W1g = W1 + (size_t)s * H1D * H2D;
  const float* W2g = W2 + (size_t)s * H2D * H3D;

  if (tid < H1D) { b0s[tid] = b0[s * H1D + tid]; b1s[tid] = b1[s * H2D + tid]; }
  if (tid < H3D) { b2s[tid] = b2[s * H3D + tid]; w3s[tid] = W3[s * H3D + tid]; }
  if (tid == 0)  { b3s = b3[s]; }

  __bf16* myAct  = actS[wave];
  float*  myActF = (float*)myAct;             // f32 view for layer-0 x staging (4 KB)
  const v8f vzero = {0.f, 0.f, 0.f, 0.f, 0.f, 0.f, 0.f, 0.f};
  v8f acc[8];
  v16bf bfr[8];

  // Per-thread K-pair decomposition of a 64-row weight chunk:
  // iteration n handles rows (2*rp, 2*rp+1) at column c; packed bf16 pair is
  // exactly the [n][k] layout's 32-bit unit -> one v_cvt_pk + one ds_store_b32.
  const int tidc  = tid & (H1D - 1);          // column for H=128 staging
  const int tidrp = tid >> 7;                 // row-pair base for H=128 staging

  // ---------------- Layer 0: [16x384] @ [384x128] ----------------
#pragma unroll
  for (int nt = 0; nt < 8; ++nt) acc[nt] = vzero;

  const float* xg0 = x + ((size_t)bRow0 * NATOMS + atom) * DIN;

  // software pipeline: next chunk's weights held in registers during compute
  float2 wr[16];
#pragma unroll
  for (int n = 0; n < 16; ++n) {              // prologue: chunk 0
    int rp = tidrp + n * 2;                   // (tid + n*256)>>7
    const float* p = W0g + (size_t)(2 * rp) * H1D + tidc;
    wr[n].x = p[0];
    wr[n].y = p[H1D];
  }

  for (int kc = 0; kc < DIN / 64; ++kc) {
    __syncthreads();                          // all waves done reading wbufT
#pragma unroll
    for (int n = 0; n < 16; ++n) {            // cvt+store staged chunk (b32 packed)
      int rp = tidrp + n * 2;
      v2bf pk = { (__bf16)wr[n].x, (__bf16)wr[n].y };
      *(v2bf*)(wbufT + tidc * 64 + 2 * rp) = pk;
    }
    const float* xg = xg0 + kc * 64;
#ifdef HAVE_TDM
    // async 2D tile DMA into this wave's LDS region; overlaps everything below
    tdm_load_x_tile(xg, myActF);
#else
    for (int i = lane; i < 16 * 64; i += 32) {
      int m = i >> 6, c = i & 63;
      myActF[m * 64 + c] =
          __builtin_nontemporal_load(xg + (size_t)m * (NATOMS * DIN) + c);
    }
#endif
    __syncthreads();                          // wbufT visible to all waves
    if (kc + 1 < DIN / 64) {                  // prefetch next chunk during WMMA
#pragma unroll
      for (int n = 0; n < 16; ++n) {
        int rp = tidrp + n * 2;
        const float* p = W0g + (size_t)((kc + 1) * 64 + 2 * rp) * H1D + tidc;
        wr[n].x = p[0];
        wr[n].y = p[H1D];
      }
    }
#ifdef HAVE_TDM
    __builtin_amdgcn_s_wait_tensorcnt((short)0);
#endif
#pragma unroll
    for (int ks = 0; ks < 2; ++ks) {
      v16bf a = load_frag_f32(myActF, mr, 64, ks * 32, h);
#pragma unroll
      for (int nt = 0; nt < 8; ++nt)   // preload B frags -> batched dscnt waits
        bfr[nt] = load_frag_bf16(wbufT + nt * (16 * 64), mr, 64, ks * 32, h);
#pragma unroll
      for (int nt = 0; nt < 8; ++nt)
        acc[nt] = __builtin_amdgcn_wmma_f32_16x16x32_bf16(
            false, a, false, bfr[nt], (short)0, acc[nt], false, false);
    }
  }
  // bias + gaussian activation, store D-frag -> act LDS [16][128] (bf16)
#pragma unroll
  for (int nt = 0; nt < 8; ++nt) {
    int nc = nt * 16 + mr;
    float bv = b0s[nc];
#pragma unroll
    for (int v = 0; v < 8; ++v) {
      float y = acc[nt][v] + bv;
      y = __expf(-y * y);
      myAct[(v + 8 * h) * H1D + nc] = (__bf16)y;
    }
  }

  // ---------------- Layer 1: [16x128] @ [128x128] ----------------
#pragma unroll
  for (int nt = 0; nt < 8; ++nt) acc[nt] = vzero;
  for (int kc = 0; kc < 2; ++kc) {
    __syncthreads();
#pragma unroll
    for (int n = 0; n < 16; ++n) {
      int rp = tidrp + n * 2;
      const float* p = W1g + (size_t)(kc * 64 + 2 * rp) * H2D + tidc;
      v2bf pk = { (__bf16)p[0], (__bf16)p[H2D] };
      *(v2bf*)(wbufT + tidc * 64 + 2 * rp) = pk;
    }
    __syncthreads();
#pragma unroll
    for (int ks = 0; ks < 2; ++ks) {
      v16bf a = load_frag_bf16(myAct, mr, H1D, kc * 64 + ks * 32, h);
#pragma unroll
      for (int nt = 0; nt < 8; ++nt)
        bfr[nt] = load_frag_bf16(wbufT + nt * (16 * 64), mr, 64, ks * 32, h);
#pragma unroll
      for (int nt = 0; nt < 8; ++nt)
        acc[nt] = __builtin_amdgcn_wmma_f32_16x16x32_bf16(
            false, a, false, bfr[nt], (short)0, acc[nt], false, false);
    }
  }
#pragma unroll
  for (int nt = 0; nt < 8; ++nt) {
    int nc = nt * 16 + mr;
    float bv = b1s[nc];
#pragma unroll
    for (int v = 0; v < 8; ++v) {
      float y = acc[nt][v] + bv;
      y = __expf(-y * y);
      myAct[(v + 8 * h) * H2D + nc] = (__bf16)y;
    }
  }

  // ---------------- Layer 2: [16x128] @ [128x64] ----------------
  const int tidc2  = tid & (H3D - 1);         // column for H=64 staging
  const int tidrp2 = tid >> 6;                // row-pair base for H=64 staging
#pragma unroll
  for (int nt = 0; nt < 4; ++nt) acc[nt] = vzero;
  for (int kc = 0; kc < 2; ++kc) {
    __syncthreads();
#pragma unroll
    for (int n = 0; n < 8; ++n) {
      int rp = tidrp2 + n * 4;
      const float* p = W2g + (size_t)(kc * 64 + 2 * rp) * H3D + tidc2;
      v2bf pk = { (__bf16)p[0], (__bf16)p[H3D] };
      *(v2bf*)(wbufT + tidc2 * 64 + 2 * rp) = pk;
    }
    __syncthreads();
#pragma unroll
    for (int ks = 0; ks < 2; ++ks) {
      v16bf a = load_frag_bf16(myAct, mr, H2D, kc * 64 + ks * 32, h);
#pragma unroll
      for (int nt = 0; nt < 4; ++nt)
        bfr[nt] = load_frag_bf16(wbufT + nt * (16 * 64), mr, 64, ks * 32, h);
#pragma unroll
      for (int nt = 0; nt < 4; ++nt)
        acc[nt] = __builtin_amdgcn_wmma_f32_16x16x32_bf16(
            false, a, false, bfr[nt], (short)0, acc[nt], false, false);
    }
  }
#pragma unroll
  for (int nt = 0; nt < 4; ++nt) {
    int nc = nt * 16 + mr;
    float bv = b2s[nc];
#pragma unroll
    for (int v = 0; v < 8; ++v) {
      float y = acc[nt][v] + bv;
      y = __expf(-y * y);
      myAct[(v + 8 * h) * H3D + nc] = (__bf16)y;
    }
  }

  // ---------------- Layer 3: [16x64] @ [64x1] + atom reduction ----------------
  if (lane < 16) {
    float sum = b3s;
#pragma unroll
    for (int k = 0; k < H3D; ++k)
      sum += (float)myAct[lane * H3D + k] * w3s[k];
    atomicAdd(out + bRow0 + lane, sum);
  }
}

extern "C" void kernel_launch(void* const* d_in, const int* in_sizes, int n_in,
                              void* d_out, int out_size, void* d_ws, size_t ws_size,
                              hipStream_t stream) {
  const float* x       = (const float*)d_in[0];
  const int*   species = (const int*)  d_in[1];
  const float* W0      = (const float*)d_in[2];
  const float* b0      = (const float*)d_in[3];
  const float* W1      = (const float*)d_in[4];
  const float* b1      = (const float*)d_in[5];
  const float* W2      = (const float*)d_in[6];
  const float* b2      = (const float*)d_in[7];
  const float* W3      = (const float*)d_in[8];
  const float* b3      = (const float*)d_in[9];
  float* out = (float*)d_out;

  (void)hipMemsetAsync(out, 0, (size_t)out_size * sizeof(float), stream);

  dim3 grid((NBATCH / 128) * NATOMS);   // 16 batch tiles x 64 atoms = 1024 blocks
  species_mlp_kernel<<<grid, 256, 0, stream>>>(x, species, W0, b0, W1, b1,
                                               W2, b2, W3, b3, out);
}
